// Attention_22016002359668
// MI455X (gfx1250) — matrix operand order repaired and checked by reading, not compiled, not run on hardware
//
#include <hip/hip_runtime.h>

typedef unsigned short u16;
typedef unsigned int   u32;

typedef __attribute__((ext_vector_type(16))) __bf16 v16bf;
typedef __attribute__((ext_vector_type(8)))  float  v8f;

static constexpr int AB  = 2;     // batch
static constexpr int AS  = 2048;  // seq
static constexpr int AD  = 1024;  // model dim
static constexpr int AH  = 16;    // heads
static constexpr int ADH = 64;    // head dim

union FragBf {
  v16bf bf;
  uint4 q[2];
  u16   h[16];
};

__device__ __forceinline__ u16 f2bf(float x) {
  u32 u = __float_as_uint(x);
  u32 r = (u + 0x7FFFu + ((u >> 16) & 1u)) >> 16;  // round-nearest-even
  return (u16)r;
}

__device__ __forceinline__ v8f wmma_bf16(v16bf a, v16bf b, v8f c) {
  // D = A(16x32 bf16) * B(32x16 bf16) + C(16x16 f32)
  return __builtin_amdgcn_wmma_f32_16x16x32_bf16(false, a, false, b, (short)0, c,
                                                 false, false);
}

// A-fragment (16x32, 16-bit): lane m=lane&15 holds row m; half g=lane>>4 selects
// K chunks [g*8 .. g*8+7] and [16+g*8 .. 16+g*8+7]  -> two 16B contiguous loads.
__device__ __forceinline__ v16bf load_a(const u16* p, int ld, int row0, int col0) {
  int lane = threadIdx.x & 31;
  int m = lane & 15, g = lane >> 4;
  const u16* r = p + (size_t)(row0 + m) * ld + col0 + g * 8;
  FragBf f;
  f.q[0] = *(const uint4*)(r);
  f.q[1] = *(const uint4*)(r + 16);
  return f.bf;
}

// B-fragment (32x16, 16-bit): lane n=lane&15 holds column n; half g=lane>>4 selects
// K rows [g*8 .. g*8+7] and [16+g*8 .. 16+g*8+7] (same order as load_a).
__device__ __forceinline__ v16bf load_b(const u16* p, int ld, int row0, int col0) {
  int lane = threadIdx.x & 31;
  int n = lane & 15, g = lane >> 4;
  const u16* r = p + (size_t)(row0 + g * 8) * ld + col0 + n;
  FragBf f;
#pragma unroll
  for (int s = 0; s < 8; ++s) {
    f.h[s]     = r[(size_t)s * ld];
    f.h[8 + s] = r[(size_t)(16 + s) * ld];
  }
  return f.bf;
}

// ---------------- prep kernels ----------------
__global__ void k_cvt_f32_bf16(const float* __restrict__ src, u16* __restrict__ dst,
                               int n) {
  int i = blockIdx.x * blockDim.x + threadIdx.x;
  int stride = gridDim.x * blockDim.x;
  for (; i < n; i += stride) dst[i] = f2bf(src[i]);
}

// W[h][n][k] (fp32) -> WT[h][k][n] (bf16)
__global__ void k_transpose_w(const float* __restrict__ W, u16* __restrict__ WT,
                              int nh, int nn, int nk) {
  int total = nh * nn * nk;
  int i = blockIdx.x * blockDim.x + threadIdx.x;
  int stride = gridDim.x * blockDim.x;
  for (; i < total; i += stride) {
    int h   = i / (nk * nn);
    int rem = i - h * nk * nn;
    int k   = rem / nn;
    int n   = rem - k * nn;
    WT[i] = f2bf(W[((size_t)h * nn + n) * nk + k]);
  }
}

// ---------------- QKV projection ----------------
// Wave computes a 16(seq) x 64(dh) tile for one of {Q,K,V} of one (b,h).
// Q,V stored [bh][S][DH]; K stored transposed [bh][DH][S] for the score GEMM.
__global__ void k_qkv(const u16* __restrict__ xb,
                      const u16* __restrict__ WqT, const u16* __restrict__ WkT,
                      const u16* __restrict__ WvT,
                      const float* __restrict__ bQ, const float* __restrict__ bK,
                      const float* __restrict__ bV,
                      u16* __restrict__ Q, u16* __restrict__ Kt,
                      u16* __restrict__ V) {
  int wid  = threadIdx.x >> 5;
  int lane = threadIdx.x & 31;
  int s0   = (blockIdx.x * 4 + wid) * 16;
  int bh   = blockIdx.y;
  int b    = bh / AH, h = bh % AH;
  int which = blockIdx.z;

  const u16*   W    = (which == 0) ? WqT : (which == 1) ? WkT : WvT;
  const float* bias = (which == 0) ? bQ : (which == 1) ? bK : bV;
  W    += (size_t)h * AD * ADH;
  bias += h * ADH;
  const u16* A = xb + ((size_t)b * AS + s0) * AD;

  v8f acc[4] = {};
  for (int k0 = 0; k0 < AD; k0 += 32) {
    v16bf a = load_a(A, AD, 0, k0);
#pragma unroll
    for (int t = 0; t < 4; ++t) {
      v16bf bm = load_b(W, ADH, k0, t * 16);
      acc[t] = wmma_bf16(a, bm, acc[t]);
    }
  }

  int laneN = lane & 15, g = lane >> 4;
  if (which != 1) {  // Q or V, row-major [S][DH]
    u16* dst = ((which == 0) ? Q : V) + ((size_t)bh * AS + s0) * ADH;
#pragma unroll
    for (int t = 0; t < 4; ++t) {
      float bi = bias[t * 16 + laneN];
#pragma unroll
      for (int r = 0; r < 8; ++r)
        dst[(size_t)(g * 8 + r) * ADH + t * 16 + laneN] = f2bf(acc[t][r] + bi);
    }
  } else {  // K transposed [DH][S]: per-lane rows are contiguous seq -> b128 store
#pragma unroll
    for (int t = 0; t < 4; ++t) {
      int n = t * 16 + laneN;
      float bi = bias[n];
      u32 w[4];
#pragma unroll
      for (int rp = 0; rp < 4; ++rp) {
        u32 lo = f2bf(acc[t][2 * rp] + bi);
        u32 hi = f2bf(acc[t][2 * rp + 1] + bi);
        w[rp]  = lo | (hi << 16);
      }
      uint4 pack; pack.x = w[0]; pack.y = w[1]; pack.z = w[2]; pack.w = w[3];
      u16* dst = Kt + ((size_t)bh * ADH + n) * AS + s0 + g * 8;
      *(uint4*)dst = pack;
    }
  }
}

// ---------------- flash attention ----------------
// Wave owns one 16-query tile; streams 32-key blocks with online softmax.
// C-fragment rows live in 16-lane halves, so shfl_xor 1/2/4/8 = per-row reduce.
__global__ void k_attn(const u16* __restrict__ Q, const u16* __restrict__ Kt,
                       const u16* __restrict__ V, u16* __restrict__ Z) {
  __shared__ u16 lds_p[4][16 * 32];
  int wid  = threadIdx.x >> 5;
  int lane = threadIdx.x & 31;
  int laneN = lane & 15, g = lane >> 4;
  int q0 = (blockIdx.x * 4 + wid) * 16;
  int bh = blockIdx.y;
  int b  = bh >> 4, h = bh & 15;

  const u16* Qb = Q + ((size_t)bh * AS + q0) * ADH;
  const u16* Kb = Kt + (size_t)bh * ADH * AS;
  const u16* Vb = V + (size_t)bh * AS * ADH;

  v16bf qa0 = load_a(Qb, ADH, 0, 0);
  v16bf qa1 = load_a(Qb, ADH, 0, 32);

  v8f o[4] = {};
  float mrow[8], lrow[8];
#pragma unroll
  for (int r = 0; r < 8; ++r) { mrow[r] = -3.0e38f; lrow[r] = 0.0f; }

  const float inv_scale = 0.125f;  // 1/sqrt(64)
  int nblk = (q0 + 16 + 31) >> 5;  // causal: keys 0 .. q0+15
  u16* lp = &lds_p[wid][0];

  for (int blk = 0; blk < nblk; ++blk) {
    int k0 = blk * 32;
    v8f sc0 = {}, sc1 = {};
    sc0 = wmma_bf16(qa0, load_b(Kb, AS, 0,  k0),      sc0);
    sc0 = wmma_bf16(qa1, load_b(Kb, AS, 32, k0),      sc0);
    sc1 = wmma_bf16(qa0, load_b(Kb, AS, 0,  k0 + 16), sc1);
    sc1 = wmma_bf16(qa1, load_b(Kb, AS, 32, k0 + 16), sc1);

    bool boundary = (k0 + 31 > q0);
    int qbase = q0 + g * 8;
#pragma unroll
    for (int r = 0; r < 8; ++r) {
      float a = sc0[r] * inv_scale;
      float c = sc1[r] * inv_scale;
      if (boundary) {
        if (k0 + laneN > qbase + r)      a = -1.0e30f;
        if (k0 + 16 + laneN > qbase + r) c = -1.0e30f;
      }
      sc0[r] = a; sc1[r] = c;
    }

#pragma unroll
    for (int r = 0; r < 8; ++r) {
      float v = fmaxf(sc0[r], sc1[r]);
      v = fmaxf(v, __shfl_xor(v, 1));
      v = fmaxf(v, __shfl_xor(v, 2));
      v = fmaxf(v, __shfl_xor(v, 4));
      v = fmaxf(v, __shfl_xor(v, 8));
      float mnew = fmaxf(mrow[r], v);
      float corr = __expf(mrow[r] - mnew);
      float p0 = __expf(sc0[r] - mnew);
      float p1 = __expf(sc1[r] - mnew);
      sc0[r] = p0; sc1[r] = p1;
      float rs = p0 + p1;
      rs += __shfl_xor(rs, 1);
      rs += __shfl_xor(rs, 2);
      rs += __shfl_xor(rs, 4);
      rs += __shfl_xor(rs, 8);
      lrow[r] = lrow[r] * corr + rs;
      mrow[r] = mnew;
#pragma unroll
      for (int t = 0; t < 4; ++t) o[t][r] *= corr;
    }

    // P (C layout, f32) -> LDS 16x32 bf16 tile -> reload as A fragment
#pragma unroll
    for (int r = 0; r < 8; ++r) {
      lp[(g * 8 + r) * 32 + laneN]      = f2bf(sc0[r]);
      lp[(g * 8 + r) * 32 + 16 + laneN] = f2bf(sc1[r]);
    }
    asm volatile("s_wait_dscnt 0" ::: "memory");
    v16bf pa = load_a(lp, 32, 0, 0);
#pragma unroll
    for (int t = 0; t < 4; ++t) {
      v16bf vb = load_b(Vb, ADH, k0, t * 16);
      o[t] = wmma_bf16(pa, vb, o[t]);
    }
  }

  // normalize and scatter into Z[b][s][h*64 + dh] (head-concatenated)
  u16* Zb = Z + ((size_t)b * AS + q0) * (AH * ADH) + h * ADH;
#pragma unroll
  for (int r = 0; r < 8; ++r) {
    float inv = 1.0f / lrow[r];
#pragma unroll
    for (int t = 0; t < 4; ++t)
      Zb[(size_t)(g * 8 + r) * (AH * ADH) + t * 16 + laneN] = f2bf(o[t][r] * inv);
  }
}

// ---------------- output projection ----------------
__global__ void k_outproj(const u16* __restrict__ Z, const u16* __restrict__ WoT,
                          const float* __restrict__ bO, float* __restrict__ out) {
  int wid  = threadIdx.x >> 5;
  int lane = threadIdx.x & 31;
  int laneN = lane & 15, g = lane >> 4;
  int row0 = (blockIdx.x * 4 + wid) * 16;
  int n0   = blockIdx.y * 64;

  v8f acc[4] = {};
  for (int k0 = 0; k0 < AH * ADH; k0 += 32) {
    v16bf a = load_a(Z, AH * ADH, row0, k0);
#pragma unroll
    for (int t = 0; t < 4; ++t) {
      v16bf bm = load_b(WoT, AD, k0, n0 + t * 16);
      acc[t] = wmma_bf16(a, bm, acc[t]);
    }
  }
  float* dst = out + (size_t)row0 * AD + n0;
#pragma unroll
  for (int t = 0; t < 4; ++t) {
    float bi = bO[n0 + t * 16 + laneN];
#pragma unroll
    for (int r = 0; r < 8; ++r)
      dst[(size_t)(g * 8 + r) * AD + t * 16 + laneN] = acc[t][r] + bi;
  }
}

extern "C" void kernel_launch(void* const* d_in, const int* in_sizes, int n_in,
                              void* d_out, int out_size, void* d_ws, size_t ws_size,
                              hipStream_t stream) {
  (void)in_sizes; (void)n_in; (void)out_size; (void)ws_size;
  const float* x  = (const float*)d_in[0];
  const float* WQ = (const float*)d_in[1];
  const float* WK = (const float*)d_in[2];
  const float* WV = (const float*)d_in[3];
  const float* WO = (const float*)d_in[4];
  const float* bQ = (const float*)d_in[5];
  const float* bK = (const float*)d_in[6];
  const float* bV = (const float*)d_in[7];
  const float* bO = (const float*)d_in[8];
  float* out = (float*)d_out;

  char* ws = (char*)d_ws;
  size_t off = 0;
  auto carve = [&](size_t elems) {
    u16* p = (u16*)(ws + off);
    off += ((elems * sizeof(u16)) + 255) & ~(size_t)255;
    return p;
  };
  u16* xb  = carve((size_t)AB * AS * AD);        // x in bf16
  u16* WqT = carve((size_t)AH * AD * ADH);       // [H][D][DH]
  u16* WkT = carve((size_t)AH * AD * ADH);
  u16* WvT = carve((size_t)AH * AD * ADH);
  u16* WoT = carve((size_t)AH * ADH * AD);       // [H*DH][D]
  u16* Qb  = carve((size_t)AB * AH * AS * ADH);  // [bh][S][DH]
  u16* Ktb = carve((size_t)AB * AH * ADH * AS);  // [bh][DH][S]
  u16* Vb  = carve((size_t)AB * AH * AS * ADH);  // [bh][S][DH]
  u16* Zb  = carve((size_t)AB * AS * AH * ADH);  // [B*S][H*DH]

  k_cvt_f32_bf16<<<4096, 256, 0, stream>>>(x, xb, AB * AS * AD);
  k_transpose_w<<<2048, 256, 0, stream>>>(WQ, WqT, AH, ADH, AD);
  k_transpose_w<<<2048, 256, 0, stream>>>(WK, WkT, AH, ADH, AD);
  k_transpose_w<<<2048, 256, 0, stream>>>(WV, WvT, AH, ADH, AD);
  k_transpose_w<<<2048, 256, 0, stream>>>(WO, WoT, AH, AD, ADH);

  k_qkv<<<dim3(AS / 64, AB * AH, 3), 128, 0, stream>>>(
      xb, WqT, WkT, WvT, bQ, bK, bV, Qb, Ktb, Vb);

  k_attn<<<dim3(AS / 64, AB * AH), 128, 0, stream>>>(Qb, Ktb, Vb, Zb);

  k_outproj<<<dim3((AB * AS) / 64, AD / 64), 128, 0, stream>>>(Zb, WoT, bO, out);
}
